// SE3TransformerLayer_58523224375461
// MI455X (gfx1250) — matrix operand order repaired and checked by reading, not compiled, not run on hardware
//
#include <hip/hip_runtime.h>
#include <cstdint>
#include <cstddef>

typedef float v8f  __attribute__((ext_vector_type(8)));
typedef float v4f  __attribute__((ext_vector_type(4)));
typedef __bf16 v16bf __attribute__((ext_vector_type(16)));
typedef unsigned int u32;

// Problem constants (from reference)
#define S_   32
#define V_   16
#define H_   8
#define NRB_ 16
#define RH_  64
#define TH_  80          // S + 3V
#define WNUM_ 2560
#define NT_  160         // WNUM / 16 column tiles
#define KCH_ 2           // two K=32 chunks of W2 (bias handled via acc init)

// coefficient LDS layout offsets (floats, per wave)
#define OXA  0           // xA[s][e]         32*16
#define ODOT 512         // dot[v][e]        16*16
#define OSS  768         // s_src[s][e]      32*16
#define OVY  1280        // vy0[v*3+c][e]    48*16
#define OCR  2048        // crs[v*3+c][e]    48*16
#define OY1  2816        // y1[c][e]          3*16
#define COFN 2880        // padded stride

__device__ __forceinline__ unsigned short f2bf(float f) {
  u32 u = __float_as_uint(f);
  u32 r = u + 0x7fffu + ((u >> 16) & 1u);
  return (unsigned short)(r >> 16);
}

union BfVec { u32 u[8]; v16bf v; };

struct BTile { uint4 a, b, c, d; };  // one 16-col tile: two K=32 B-fragments

__device__ __forceinline__ void loadB(BTile& bt, const uint4* bbase, int t, int lane) {
  const uint4* bp = bbase + ((size_t)(t * KCH_) * 32 + lane) * 2;
  bt.a = bp[0];  bt.b = bp[1];    // K chunk 0
  bt.c = bp[64]; bt.d = bp[65];   // K chunk 1
}

__device__ __forceinline__ v8f wmma2(const BfVec& a0, const BfVec& a1,
                                     const BTile& bt, float bias) {
  BfVec b0, b1;
  b0.u[0] = bt.a.x; b0.u[1] = bt.a.y; b0.u[2] = bt.a.z; b0.u[3] = bt.a.w;
  b0.u[4] = bt.b.x; b0.u[5] = bt.b.y; b0.u[6] = bt.b.z; b0.u[7] = bt.b.w;
  b1.u[0] = bt.c.x; b1.u[1] = bt.c.y; b1.u[2] = bt.c.z; b1.u[3] = bt.c.w;
  b1.u[4] = bt.d.x; b1.u[5] = bt.d.y; b1.u[6] = bt.d.z; b1.u[7] = bt.d.w;
  v8f acc = {bias, bias, bias, bias, bias, bias, bias, bias};  // b2 folded here
  acc = __builtin_amdgcn_wmma_f32_16x16x32_bf16(false, a0.v, false, b0.v, (short)0, acc, false, false);
  acc = __builtin_amdgcn_wmma_f32_16x16x32_bf16(false, a1.v, false, b1.v, (short)0, acc, false, false);
  return acc;
}

__device__ __forceinline__ void fma8(float* dst, const float* p, const v8f& acc) {
  v4f f0 = *(const v4f*)p;
  v4f f1 = *(const v4f*)(p + 4);
#pragma unroll
  for (int r = 0; r < 8; ++r) dst[r] += ((r < 4) ? f0[r] : f1[r - 4]) * acc[r];
}

// -------------------------------------------------------------------------
// Prep: repack W2 (fp32 64x2560) into bf16 WMMA B-fragment layout.
// frag[((t*2+c)*32 + lane)*8 .. +7] : lane's 16 bf16 K-values for column
// N = t*16 + lane%16, K = c*32 + 16*(lane/16) + i
// -------------------------------------------------------------------------
__global__ void k_prep_w2frag(const float* __restrict__ W2, u32* __restrict__ frag) {
  int id = blockIdx.x * blockDim.x + threadIdx.x;
  if (id >= NT_ * KCH_ * 32) return;
  int lane = id & 31;
  int tc = id >> 5;
  int c = tc & (KCH_ - 1);
  int t = tc >> 1;
  int ncol = t * 16 + (lane & 15);
  int kbase = c * 32 + ((lane >> 4) << 4);
  u32* dst = frag + (size_t)id * 8;
#pragma unroll
  for (int w = 0; w < 8; ++w) {
    float f0 = W2[(size_t)(kbase + 2 * w) * WNUM_ + ncol];
    float f1 = W2[(size_t)(kbase + 2 * w + 1) * WNUM_ + ncol];
    dst[w] = (u32)f2bf(f0) | ((u32)f2bf(f1) << 16);
  }
}

// -------------------------------------------------------------------------
// Init accumulators / encoded segment-max
// -------------------------------------------------------------------------
__global__ void k_init(float* __restrict__ den, u32* __restrict__ maxEnc,
                       float* __restrict__ agg, float* __restrict__ stats, int N) {
  int id = blockIdx.x * blockDim.x + threadIdx.x;
  if (id < N * TH_) agg[id] = 0.f;
  if (id < N) { den[id] = 0.f; maxEnc[id] = 0x007FFFFFu; }  // enc(-inf)
  if (id < 80) stats[id] = 0.f;
}

// -------------------------------------------------------------------------
// Fused edge kernel: radial MLP (bf16 WMMA, b2 folded into acc init),
// per-edge tensor-product contraction streamed over 80 tile-pairs with
// register-double-buffered B fragments, message assembly, attention logits.
// Block = 128 threads = 4 waves; each wave owns 16 edges.
// -------------------------------------------------------------------------
__global__ __launch_bounds__(128) void k_edges(
    const float* __restrict__ nf, const int* __restrict__ eidx,
    const float* __restrict__ esh, const float* __restrict__ erad,
    const float* __restrict__ eenv,
    const float* __restrict__ W1, const float* __restrict__ b1,
    const float* __restrict__ b2,
    const float* __restrict__ Wq, const float* __restrict__ bq,
    const float* __restrict__ Wk, const float* __restrict__ bk,
    const u32* __restrict__ w2frag,
    float* __restrict__ msg, float* __restrict__ logits, int E) {
  __shared__ float sW1[NRB_ * RH_];
  __shared__ float sb1[RH_];
  __shared__ float sWq[S_ * H_];
  __shared__ float sbq[H_];
  __shared__ float sWk[S_ * H_];
  __shared__ float sbk[H_];
  __shared__ __align__(16) unsigned short hb[4][16][RH_];  // hidden, bf16 bits
  __shared__ __align__(16) float cof[4][COFN];             // coeffs; [0,1280) reused as msgbuf

  int tid = threadIdx.x;
  for (int i = tid; i < NRB_ * RH_; i += 128) sW1[i] = W1[i];
  for (int i = tid; i < RH_; i += 128) sb1[i] = b1[i];
  for (int i = tid; i < S_ * H_; i += 128) { sWq[i] = Wq[i]; sWk[i] = Wk[i]; }
  for (int i = tid; i < H_; i += 128) { sbq[i] = bq[i]; sbk[i] = bk[i]; }
  __syncthreads();

  int w = tid >> 5, lane = tid & 31;
  int n = lane & 15, hi = lane >> 4, e0 = hi * 8;
  int edgeBase = blockIdx.x * 64 + w * 16;
  float* cw = cof[w];

  // ---- hidden layer h = silu(radial @ W1 + b1): lane -> (edge = lane%16, j half = hi)
  {
    int e = lane & 15;
    int jb = hi * 32;
    int eg = edgeBase + e;
    float rad[16];
#pragma unroll
    for (int r = 0; r < 16; ++r) rad[r] = 0.f;
    if (eg < E) {
      const float* rp = erad + (size_t)eg * NRB_;
#pragma unroll
      for (int r = 0; r < 16; ++r) rad[r] = rp[r];
    }
    for (int j = 0; j < 32; ++j) {
      float a = sb1[jb + j];
#pragma unroll
      for (int r = 0; r < 16; ++r) a += rad[r] * sW1[r * RH_ + jb + j];
      float hsig = a / (1.f + __expf(-a));
      hb[w][e][jb + j] = f2bf(hsig);
    }
  }

  // ---- per-edge geometric coefficients (lanes 0..15, one edge each)
  if (lane < 16) {
    int e = lane;
    int eg = edgeBase + e;
    const float i3 = 0.57735026919f, i2 = 0.70710678119f;
    if (eg < E) {
      const float* shp = esh + (size_t)eg * 4;
      float y0 = shp[0], y1x = shp[1], y1y = shp[2], y1z = shp[3];
      const float* np = nf + (size_t)eidx[eg] * TH_;
#pragma unroll
      for (int s = 0; s < S_; ++s) {
        float ss = np[s];
        cw[OSS + s * 16 + e] = ss;
        cw[OXA + s * 16 + e] = ss * y0;
      }
#pragma unroll
      for (int v = 0; v < V_; ++v) {
        float vx = np[S_ + v * 3 + 0], vy = np[S_ + v * 3 + 1], vz = np[S_ + v * 3 + 2];
        cw[ODOT + v * 16 + e] = (vx * y1x + vy * y1y + vz * y1z) * i3;
        cw[OVY + (v * 3 + 0) * 16 + e] = vx * y0;
        cw[OVY + (v * 3 + 1) * 16 + e] = vy * y0;
        cw[OVY + (v * 3 + 2) * 16 + e] = vz * y0;
        cw[OCR + (v * 3 + 0) * 16 + e] = (vy * y1z - vz * y1y) * i2;
        cw[OCR + (v * 3 + 1) * 16 + e] = (vz * y1x - vx * y1z) * i2;
        cw[OCR + (v * 3 + 2) * 16 + e] = (vx * y1y - vy * y1x) * i2;
      }
      cw[OY1 + 0 * 16 + e] = y1x; cw[OY1 + 1 * 16 + e] = y1y; cw[OY1 + 2 * 16 + e] = y1z;
    } else {
#pragma unroll
      for (int s = 0; s < S_; ++s) { cw[OSS + s * 16 + e] = 0.f; cw[OXA + s * 16 + e] = 0.f; }
#pragma unroll
      for (int v = 0; v < V_; ++v) {
        cw[ODOT + v * 16 + e] = 0.f;
#pragma unroll
        for (int c = 0; c < 3; ++c) {
          cw[OVY + (v * 3 + c) * 16 + e] = 0.f;
          cw[OCR + (v * 3 + c) * 16 + e] = 0.f;
        }
      }
      cw[OY1 + e] = 0.f; cw[OY1 + 16 + e] = 0.f; cw[OY1 + 32 + e] = 0.f;
    }
  }
  __syncthreads();

  // ---- build WMMA A-fragments (contiguous K order, same as w2frag: row M = lane%16,
  //      K = 16*hi+0..15 within each K=32 chunk)
  BfVec afr0, afr1;
  {
    uint4 x0 = *(const uint4*)&hb[w][n][16 * hi];
    uint4 x1 = *(const uint4*)&hb[w][n][16 * hi + 8];
    afr0.u[0] = x0.x; afr0.u[1] = x0.y; afr0.u[2] = x0.z; afr0.u[3] = x0.w;
    afr0.u[4] = x1.x; afr0.u[5] = x1.y; afr0.u[6] = x1.z; afr0.u[7] = x1.w;
    uint4 x2 = *(const uint4*)&hb[w][n][32 + 16 * hi];
    uint4 x3 = *(const uint4*)&hb[w][n][32 + 16 * hi + 8];
    afr1.u[0] = x2.x; afr1.u[1] = x2.y; afr1.u[2] = x2.z; afr1.u[3] = x2.w;
    afr1.u[4] = x3.x; afr1.u[5] = x3.y; afr1.u[6] = x3.z; afr1.u[7] = x3.w;
  }

  float msgS0[8], msgS1[8], tmpC[8], mvx[8], mvy[8], mvz[8];
#pragma unroll
  for (int r = 0; r < 8; ++r) { msgS0[r] = msgS1[r] = tmpC[r] = mvx[r] = mvy[r] = mvz[r] = 0.f; }

  // ---- stream 80 tile-pairs (tiles 2p, 2p+1): software-pipelined B loads
  const uint4* bbase = (const uint4*)w2frag;
  BTile c0, c1, n0;
  loadB(c0, bbase, 0, lane);
  float bias0 = b2[n];
  for (int p = 0; p < 80; ++p) {
    loadB(c1, bbase, 2 * p + 1, lane);                 // second tile of this pair
    float bias1 = b2[(2 * p + 1) * 16 + n];
    int tn = (p < 79) ? (2 * p + 2) : 158;             // next pair's first tile (clamped)
    loadB(n0, bbase, tn, lane);
    float nb0 = b2[tn * 16 + n];
    __builtin_prefetch((const void*)(bbase + ((size_t)(tn * KCH_) * 32 + lane) * 2 + 128), 0, 3);

    v8f acc0 = wmma2(afr0, afr1, c0, bias0);           // c0 loaded last iteration
    v8f acc1 = wmma2(afr0, afr1, c1, bias1);

    // acc0[r] = tp_w[edge e0+r][col 2p*16+n], acc1 = col (2p+1)*16+n
    if (p < 32) {                                      // WA: s = p, k = n / 16+n
      const float* cp = cw + OXA + p * 16 + e0;
      fma8(msgS0, cp, acc0);
      fma8(msgS1, cp, acc1);
    } else if (p < 48) {                               // WB: v = p-32
      const float* cp = cw + ODOT + (p - 32) * 16 + e0;
      fma8(msgS0, cp, acc0);
      fma8(msgS1, cp, acc1);
    } else if (p < 64) {                               // WC: s = 2(p-48), 2(p-48)+1
      int s = (p - 48) * 2;
      fma8(tmpC, cw + OSS + s * 16 + e0, acc0);
      fma8(tmpC, cw + OSS + (s + 1) * 16 + e0, acc1);
    } else if (p < 72) {                               // WD: v = 2(p-64), +1
      int v = (p - 64) * 2;
      fma8(mvx, cw + OVY + (v * 3 + 0) * 16 + e0, acc0);
      fma8(mvy, cw + OVY + (v * 3 + 1) * 16 + e0, acc0);
      fma8(mvz, cw + OVY + (v * 3 + 2) * 16 + e0, acc0);
      fma8(mvx, cw + OVY + (v * 3 + 3) * 16 + e0, acc1);
      fma8(mvy, cw + OVY + (v * 3 + 4) * 16 + e0, acc1);
      fma8(mvz, cw + OVY + (v * 3 + 5) * 16 + e0, acc1);
    } else {                                           // WE: v = 2(p-72), +1
      int v = (p - 72) * 2;
      fma8(mvx, cw + OCR + (v * 3 + 0) * 16 + e0, acc0);
      fma8(mvy, cw + OCR + (v * 3 + 1) * 16 + e0, acc0);
      fma8(mvz, cw + OCR + (v * 3 + 2) * 16 + e0, acc0);
      fma8(mvx, cw + OCR + (v * 3 + 3) * 16 + e0, acc1);
      fma8(mvy, cw + OCR + (v * 3 + 4) * 16 + e0, acc1);
      fma8(mvz, cw + OCR + (v * 3 + 5) * 16 + e0, acc1);
    }
    c0 = n0;
    bias0 = nb0;
  }

  // ---- finalize messages into per-wave LDS (rows: edge, 80 floats).
  // msgbuf aliases cof[w][0..1280) -- those coefficients are dead now.
  const float a0 = 0.14433756730f;  // 1/sqrt(S+V)
  const float a1 = 0.125f;          // 1/sqrt(S+2V)
#pragma unroll
  for (int r = 0; r < 8; ++r) {
    int e = e0 + r;
    float y1x = cw[OY1 + e], y1y = cw[OY1 + 16 + e], y1z = cw[OY1 + 32 + e];
    float* mrow = cw + e * TH_;
    mrow[n]      = a0 * msgS0[r];
    mrow[16 + n] = a0 * msgS1[r];
    mrow[S_ + n * 3 + 0] = a1 * (tmpC[r] * y1x + mvx[r]);
    mrow[S_ + n * 3 + 1] = a1 * (tmpC[r] * y1y + mvy[r]);
    mrow[S_ + n * 3 + 2] = a1 * (tmpC[r] * y1z + mvz[r]);
  }
  __syncthreads();

  // ---- attention logits (lane per edge)
  if (lane < 16) {
    int e = lane;
    int eg = edgeBase + e;
    if (eg < E) {
      const float* mrow = cw + e * TH_;
      float kk[H_], qq[H_];
#pragma unroll
      for (int h = 0; h < H_; ++h) { kk[h] = sbk[h]; qq[h] = sbq[h]; }
      const float* np = nf + (size_t)eidx[E + eg] * TH_;
      for (int s = 0; s < S_; ++s) {
        float ms = mrow[s], ds = np[s];
#pragma unroll
        for (int h = 0; h < H_; ++h) { kk[h] += ms * sWk[s * H_ + h]; qq[h] += ds * sWq[s * H_ + h]; }
      }
      float lg = 0.f;
#pragma unroll
      for (int h = 0; h < H_; ++h) lg += qq[h] * kk[h];
      logits[eg] = lg * 0.35355339059f + __logf(eenv[eg] + 1e-8f);
    }
  }

  // ---- coalesced message store (each lane writes 40 floats = half an edge row)
  {
    int eg = edgeBase + (lane >> 1);
    if (eg < E) {
      const v4f* srcp = (const v4f*)cw + lane * 10;
      v4f* dstp = (v4f*)(msg + (size_t)edgeBase * TH_) + lane * 10;
#pragma unroll
      for (int i = 0; i < 10; ++i) dstp[i] = srcp[i];
    }
  }
}

// -------------------------------------------------------------------------
// Softmax-over-segments + weighted aggregation
// -------------------------------------------------------------------------
__global__ void k_segmax(const float* __restrict__ logits, const int* __restrict__ eidx,
                         u32* __restrict__ maxEnc, int E) {
  int e = blockIdx.x * blockDim.x + threadIdx.x;
  if (e >= E) return;
  u32 u = __float_as_uint(logits[e]);
  u32 enc = (u & 0x80000000u) ? ~u : (u | 0x80000000u);  // order-preserving
  atomicMax(&maxEnc[eidx[E + e]], enc);
}

__global__ void k_expden(const float* __restrict__ logits, const int* __restrict__ eidx,
                         const u32* __restrict__ maxEnc, float* __restrict__ exbuf,
                         float* __restrict__ den, int E) {
  int e = blockIdx.x * blockDim.x + threadIdx.x;
  if (e >= E) return;
  int d = eidx[E + e];
  u32 enc = maxEnc[d];
  u32 u = (enc & 0x80000000u) ? (enc ^ 0x80000000u) : ~enc;
  float ex = __expf(logits[e] - __uint_as_float(u));
  exbuf[e] = ex;
  atomicAdd(&den[d], ex);
}

__global__ void k_agg(const float* __restrict__ msg, const float* __restrict__ exbuf,
                      const float* __restrict__ den, const int* __restrict__ eidx,
                      float* __restrict__ agg, int E) {
  int id = blockIdx.x * blockDim.x + threadIdx.x;
  if (id >= E * TH_) return;
  int e = id / TH_, f = id - e * TH_;
  int d = eidx[E + e];
  float attn = exbuf[e] / (den[d] + 1e-8f);
  atomicAdd(&agg[(size_t)d * TH_ + f], attn * msg[id]);
}

// -------------------------------------------------------------------------
// Node transforms (thread per node, matrices staged in LDS)
// -------------------------------------------------------------------------
__global__ void k_nodeA(const float* __restrict__ agg, const float* __restrict__ Wout_s,
                        const float* __restrict__ Wout_v, float* __restrict__ t1, int N) {
  __shared__ float sWs[S_ * S_];
  __shared__ float sWv[V_ * V_];
  for (int i = threadIdx.x; i < S_ * S_; i += blockDim.x) sWs[i] = Wout_s[i];
  for (int i = threadIdx.x; i < V_ * V_; i += blockDim.x) sWv[i] = Wout_v[i];
  __syncthreads();
  int nid = blockIdx.x * blockDim.x + threadIdx.x;
  if (nid >= N) return;
  const float* a = agg + (size_t)nid * TH_;
  float* o = t1 + (size_t)nid * TH_;
  const float iS = 0.17677669529f, iV = 0.25f;
  float as_[S_];
#pragma unroll
  for (int s = 0; s < S_; ++s) as_[s] = a[s];
  for (int k = 0; k < S_; ++k) {
    float acc = 0.f;
#pragma unroll
    for (int s = 0; s < S_; ++s) acc += as_[s] * sWs[s * S_ + k];
    o[k] = acc * iS;
  }
  float v1[V_ * 3];
#pragma unroll
  for (int i = 0; i < V_ * 3; ++i) v1[i] = 0.f;
  for (int v = 0; v < V_; ++v) {
    float vx = a[S_ + v * 3 + 0], vy = a[S_ + v * 3 + 1], vz = a[S_ + v * 3 + 2];
#pragma unroll
    for (int w2 = 0; w2 < V_; ++w2) {
      float wv = sWv[v * V_ + w2];
      v1[w2 * 3 + 0] += vx * wv; v1[w2 * 3 + 1] += vy * wv; v1[w2 * 3 + 2] += vz * wv;
    }
  }
#pragma unroll
  for (int i = 0; i < V_ * 3; ++i) o[S_ + i] = v1[i] * iV;
}

__global__ void k_nodeB(const float* __restrict__ t1, const float* __restrict__ nf,
                        const float* __restrict__ Wg_s, const float* __restrict__ Wg_v,
                        float* __restrict__ x, float* __restrict__ stats, int N) {
  __shared__ float sWgs[S_ * (S_ + V_)];
  __shared__ float sWgv[V_ * V_];
  __shared__ float red[80];
  for (int i = threadIdx.x; i < S_ * (S_ + V_); i += blockDim.x) sWgs[i] = Wg_s[i];
  for (int i = threadIdx.x; i < V_ * V_; i += blockDim.x) sWgv[i] = Wg_v[i];
  if (threadIdx.x < 80) red[threadIdx.x] = 0.f;
  __syncthreads();
  int nid = blockIdx.x * blockDim.x + threadIdx.x;
  if (nid < N) {
    const float* t = t1 + (size_t)nid * TH_;
    const float* np = nf + (size_t)nid * TH_;
    float* xo = x + (size_t)nid * TH_;
    const float iS = 0.17677669529f, iV = 0.25f;
    float s1[S_];
#pragma unroll
    for (int k = 0; k < S_; ++k) s1[k] = t[k];
    float gs[S_ + V_];
    for (int j = 0; j < S_ + V_; ++j) {
      float acc = 0.f;
#pragma unroll
      for (int k = 0; k < S_; ++k) acc += s1[k] * sWgs[k * (S_ + V_) + j];
      gs[j] = acc * iS;
    }
    float gv[V_ * 3];
#pragma unroll
    for (int i = 0; i < V_ * 3; ++i) gv[i] = 0.f;
    for (int v = 0; v < V_; ++v) {
      float vx = t[S_ + v * 3 + 0], vy = t[S_ + v * 3 + 1], vz = t[S_ + v * 3 + 2];
#pragma unroll
      for (int w2 = 0; w2 < V_; ++w2) {
        float wv = sWgv[v * V_ + w2];
        gv[w2 * 3 + 0] += vx * wv; gv[w2 * 3 + 1] += vy * wv; gv[w2 * 3 + 2] += vz * wv;
      }
    }
    for (int k = 0; k < S_; ++k) {
      float g = gs[k];
      float xs = g / (1.f + __expf(-g)) + np[k];
      xo[k] = xs;
      atomicAdd(&red[k], xs);
      atomicAdd(&red[32 + k], xs * xs);
    }
    for (int w2 = 0; w2 < V_; ++w2) {
      float gate = 1.f / (1.f + __expf(-gs[S_ + w2]));
      float x0 = gate * gv[w2 * 3 + 0] * iV + np[S_ + w2 * 3 + 0];
      float x1 = gate * gv[w2 * 3 + 1] * iV + np[S_ + w2 * 3 + 1];
      float x2 = gate * gv[w2 * 3 + 2] * iV + np[S_ + w2 * 3 + 2];
      xo[S_ + w2 * 3 + 0] = x0; xo[S_ + w2 * 3 + 1] = x1; xo[S_ + w2 * 3 + 2] = x2;
      atomicAdd(&red[64 + w2], (x0 * x0 + x1 * x1 + x2 * x2) * (1.f / 3.f));
    }
  }
  __syncthreads();
  if (threadIdx.x < 80) atomicAdd(&stats[threadIdx.x], red[threadIdx.x]);
}

__global__ void k_nodeC(const float* __restrict__ x, const float* __restrict__ stats,
                        const float* __restrict__ bn_ws, const float* __restrict__ bn_bs,
                        const float* __restrict__ bn_wv, float* __restrict__ out,
                        int N, float invN) {
  int nid = blockIdx.x * blockDim.x + threadIdx.x;
  if (nid >= N) return;
  const float* xi = x + (size_t)nid * TH_;
  float* oo = out + (size_t)nid * TH_;
#pragma unroll
  for (int k = 0; k < S_; ++k) {
    float mean = stats[k] * invN;
    float var = stats[32 + k] * invN - mean * mean;
    oo[k] = (xi[k] - mean) * rsqrtf(var + 1e-5f) * bn_ws[k] + bn_bs[k];
  }
#pragma unroll
  for (int v = 0; v < V_; ++v) {
    float vn = stats[64 + v] * invN;
    float sc = bn_wv[v] * rsqrtf(vn + 1e-5f);
    oo[S_ + v * 3 + 0] = xi[S_ + v * 3 + 0] * sc;
    oo[S_ + v * 3 + 1] = xi[S_ + v * 3 + 1] * sc;
    oo[S_ + v * 3 + 2] = xi[S_ + v * 3 + 2] * sc;
  }
}

// -------------------------------------------------------------------------
extern "C" void kernel_launch(void* const* d_in, const int* in_sizes, int n_in,
                              void* d_out, int out_size, void* d_ws, size_t ws_size,
                              hipStream_t stream) {
  const float* nf    = (const float*)d_in[0];
  const int*   eidx  = (const int*)d_in[1];
  const float* esh   = (const float*)d_in[2];
  const float* erad  = (const float*)d_in[3];
  const float* eenv  = (const float*)d_in[4];
  const float* W1    = (const float*)d_in[5];
  const float* b1    = (const float*)d_in[6];
  const float* W2    = (const float*)d_in[7];
  const float* b2    = (const float*)d_in[8];
  const float* Wq    = (const float*)d_in[9];
  const float* bq    = (const float*)d_in[10];
  const float* Wk    = (const float*)d_in[11];
  const float* bk    = (const float*)d_in[12];
  const float* Wout_s = (const float*)d_in[13];
  const float* Wout_v = (const float*)d_in[14];
  const float* Wg_s  = (const float*)d_in[15];
  const float* Wg_v  = (const float*)d_in[16];
  const float* bn_ws = (const float*)d_in[17];
  const float* bn_bs = (const float*)d_in[18];
  const float* bn_wv = (const float*)d_in[19];

  int N = in_sizes[0] / TH_;
  int E = in_sizes[1] / 2;

  char* wsp = (char*)d_ws;
  auto alloc = [&](size_t bytes) {
    char* p = wsp;
    wsp += (bytes + 255) & ~(size_t)255;
    return p;
  };
  float* msg    = (float*)alloc((size_t)E * TH_ * 4);
  float* logits = (float*)alloc((size_t)E * 4);
  float* exbuf  = (float*)alloc((size_t)E * 4);
  u32*   maxEnc = (u32*)alloc((size_t)N * 4);
  float* den    = (float*)alloc((size_t)N * 4);
  float* agg    = (float*)alloc((size_t)N * TH_ * 4);
  float* t1     = (float*)alloc((size_t)N * TH_ * 4);
  float* xbuf   = (float*)alloc((size_t)N * TH_ * 4);
  float* stats  = (float*)alloc(80 * 4);
  u32*   w2frag = (u32*)alloc((size_t)NT_ * KCH_ * 32 * 8 * 4);
  (void)ws_size; (void)n_in; (void)out_size;

  int thr = 256;
  k_prep_w2frag<<<(NT_ * KCH_ * 32 + thr - 1) / thr, thr, 0, stream>>>(W2, w2frag);
  k_init<<<(N * TH_ + thr - 1) / thr, thr, 0, stream>>>(den, maxEnc, agg, stats, N);
  k_edges<<<(E + 63) / 64, 128, 0, stream>>>(nf, eidx, esh, erad, eenv,
                                             W1, b1, b2, Wq, bq, Wk, bk,
                                             w2frag, msg, logits, E);
  k_segmax<<<(E + thr - 1) / thr, thr, 0, stream>>>(logits, eidx, maxEnc, E);
  k_expden<<<(E + thr - 1) / thr, thr, 0, stream>>>(logits, eidx, maxEnc, exbuf, den, E);
  k_agg<<<((size_t)E * TH_ + thr - 1) / thr, thr, 0, stream>>>(msg, exbuf, den, eidx, agg, E);
  k_nodeA<<<(N + thr - 1) / thr, thr, 0, stream>>>(agg, Wout_s, Wout_v, t1, N);
  k_nodeB<<<(N + thr - 1) / thr, thr, 0, stream>>>(t1, nf, Wg_s, Wg_v, xbuf, stats, N);
  k_nodeC<<<(N + thr - 1) / thr, thr, 0, stream>>>(xbuf, stats, bn_ws, bn_bs, bn_wv,
                                                   (float*)d_out, N, 1.f / (float)N);
}
